// SelfAttention_16363825398359
// MI455X (gfx1250) — matrix operand order repaired and checked by reading, not compiled, not run on hardware
//
#include <hip/hip_runtime.h>
#include <math.h>

// ---------------------------------------------------------------------------
// Self-attention for MI455X (gfx1250, wave32, WMMA).
//   B=4, S=2048, d_model = qk = out = 1024.
//   Phase 1: Q/K/V projections via V_WMMA_F32_16X16X32_F16 (f16 in, f32 acc).
//            V stored TRANSPOSED [b][col][key] so the PV B-operand is
//            contiguous in the key dimension.
//   Phase 2: flash-attention (online softmax); K blocks staged into LDS via
//            GLOBAL_LOAD_ASYNC_TO_LDS_B128 (ASYNCcnt) when available.
// ---------------------------------------------------------------------------

typedef __attribute__((ext_vector_type(16))) _Float16 v16h;
typedef __attribute__((ext_vector_type(8)))  _Float16 v8h;
typedef __attribute__((ext_vector_type(8)))  float    v8f;
typedef __attribute__((ext_vector_type(4)))  int      v4i;

// pointer-to-AS-qualified-int4 types for the async-LDS builtin
typedef __attribute__((address_space(1))) v4i* gv4i_p;   // global
typedef __attribute__((address_space(3))) v4i* lv4i_p;   // LDS

#define DMODEL 1024
#define SEQ    2048
#define NBATCH 4
#define NROWS  (NBATCH * SEQ)   // 8192

#if defined(__AMDGCN__) && __has_builtin(__builtin_amdgcn_global_load_async_to_lds_b128)
#define USE_ASYNC_LDS 1
#else
#define USE_ASYNC_LDS 0
#endif

__device__ __forceinline__ void wait_async0() {
#if defined(__AMDGCN__)
#if __has_builtin(__builtin_amdgcn_s_wait_asynccnt)
  __builtin_amdgcn_s_wait_asynccnt(0);
#else
  asm volatile("s_wait_asynccnt 0x0" ::: "memory");
#endif
#endif
}

// ---- WMMA fragment builders (CDNA5 wave32 16x16x32 f16 layouts) -----------
// A (16x32, MxK): lane L holds row M=L%16; halfs 0..7 = K base..base+7,
// halfs 8..15 = K base+16..base+23, base = (L/16)*8.
__device__ __forceinline__ v16h frag_a(const _Float16* p, int lane) {
  union { v16h v; v8h h[2]; } u;
  const int base = (lane >> 4) << 3;                 // 0 or 8
  u.h[0] = *reinterpret_cast<const v8h*>(p + base);
  u.h[1] = *reinterpret_cast<const v8h*>(p + base + 16);
  return u.v;
}
// B (32x16, KxN): lane L holds column N=L%16; halfs 0..7 = K base..base+7,
// halfs 8..15 = K base+16..base+23, base = (L/16)*8.
__device__ __forceinline__ v16h frag_b(const _Float16* p, int lane) {
  union { v16h v; v8h h[2]; } u;
  const int base = (lane >> 4) << 3;                 // 0 or 8
  u.h[0] = *reinterpret_cast<const v8h*>(p + base);
  u.h[1] = *reinterpret_cast<const v8h*>(p + base + 16);
  return u.v;
}

// ---------------------------------------------------------------------------
// Kernel 1: y = x @ W^T + b for W in {Wq,Wk,Wv}; outputs f16.
// grid = (NROWS/64, DMODEL/64, 3), block = 256 (8 waves).
// ---------------------------------------------------------------------------
#define APAD 72   // 64 + 8 halfs: bank stride 36 words -> conflict-free

__global__ __launch_bounds__(256) void qkv_gemm_kernel(
    const float* __restrict__ x,
    const float* __restrict__ Wq, const float* __restrict__ bq,
    const float* __restrict__ Wk, const float* __restrict__ bk,
    const float* __restrict__ Wv, const float* __restrict__ bv,
    _Float16* __restrict__ Qb, _Float16* __restrict__ Kb,
    _Float16* __restrict__ VbT)
{
  __shared__ __align__(16) _Float16 As[64][APAD];   // x tile   (rows x k)
  __shared__ __align__(16) _Float16 Ws[64][APAD];   // W tile   (outcol x k)

  const int tid  = threadIdx.x;
  const int lane = tid & 31;
  const int wave = tid >> 5;
  const int rowBase = blockIdx.x * 64;
  const int colBase = blockIdx.y * 64;
  const int which   = blockIdx.z;                   // 0=Q 1=K 2=V

  const float* W    = (which == 0) ? Wq : (which == 1) ? Wk : Wv;
  const float* bias = (which == 0) ? bq : (which == 1) ? bk : bv;

  v8f acc0 = {}, acc1 = {};
  const int t0 = wave * 2;                          // tiles t0, t0+1 of 16 (4x4)

  for (int d0 = 0; d0 < DMODEL; d0 += 64) {
    for (int i = 0; i < 4; ++i) {
      const int e4 = i * 256 + tid;                 // float4 index, 1024 total
      const int r  = e4 >> 4;                       // 0..63
      const int c4 = (e4 & 15) << 2;                // 0..60
      const float4 vx =
          *reinterpret_cast<const float4*>(&x[(size_t)(rowBase + r) * DMODEL + d0 + c4]);
      As[r][c4 + 0] = (_Float16)vx.x; As[r][c4 + 1] = (_Float16)vx.y;
      As[r][c4 + 2] = (_Float16)vx.z; As[r][c4 + 3] = (_Float16)vx.w;
      const float4 vw =
          *reinterpret_cast<const float4*>(&W[(size_t)(colBase + r) * DMODEL + d0 + c4]);
      Ws[r][c4 + 0] = (_Float16)vw.x; Ws[r][c4 + 1] = (_Float16)vw.y;
      Ws[r][c4 + 2] = (_Float16)vw.z; Ws[r][c4 + 3] = (_Float16)vw.w;
    }
    __syncthreads();

    for (int t = 0; t < 2; ++t) {
      const int tr = ((t0 + t) >> 2) << 4;
      const int tc = ((t0 + t) & 3) << 4;
      v8f* acc = t ? &acc1 : &acc0;
      for (int kk = 0; kk < 64; kk += 32) {
        const v16h a = frag_a(&As[tr + (lane & 15)][kk], lane);
        const v16h b = frag_b(&Ws[tc + (lane & 15)][kk], lane);
        *acc = __builtin_amdgcn_wmma_f32_16x16x32_f16(
            false, a, false, b, (short)0, *acc, false, false);
      }
    }
    __syncthreads();
  }

  // epilogue: bias add (N = lane&15 constant per lane), branch-free stores
  for (int t = 0; t < 2; ++t) {
    const int tr = ((t0 + t) >> 2) << 4;
    const int tc = ((t0 + t) & 3) << 4;
    const int Ncol  = colBase + tc + (lane & 15);
    const float bv_ = bias[Ncol];
    const int mrow0 = rowBase + tr + ((lane >> 4) << 3);   // 8-aligned group
    const v8f acc = t ? acc1 : acc0;

    _Float16* base;
    size_t    stride;
    if (which == 2) {            // transposed V: [b][col][key]
      const int b0  = mrow0 >> 11;                  // row / SEQ
      const int key = mrow0 & (SEQ - 1);
      base   = &VbT[((size_t)b0 * DMODEL + Ncol) * SEQ + key];
      stride = 1;
    } else {
      _Float16* Out = (which == 0) ? Qb : Kb;
      base   = &Out[(size_t)mrow0 * DMODEL + Ncol];
      stride = DMODEL;
    }
    for (int r = 0; r < 8; ++r)
      base[(size_t)r * stride] = (_Float16)(acc[r] + bv_);
  }
}

// ---------------------------------------------------------------------------
// Kernel 2: flash attention. grid = (SEQ/16, NBATCH), block = 256 (8 waves).
// ---------------------------------------------------------------------------
#define KPAD 1032   // 1024 + 8 halfs: row stride 516 words, conflict-free
#define PPAD 40     // 32 + 8 halfs

__global__ __launch_bounds__(256) void flash_attn_kernel(
    const _Float16* __restrict__ Qb, const _Float16* __restrict__ Kb,
    const _Float16* __restrict__ VbT, float* __restrict__ out)
{
  __shared__ __align__(16) _Float16 Ks[32][KPAD];   // staged K block (~64.5KB)
  __shared__ __align__(16) float    Ss[16][32];     // score reduction tile
  __shared__ __align__(16) _Float16 P[16][PPAD];    // probabilities (f16)
  __shared__ float mrow[16], lrow[16], arow[16];

  const int tid  = threadIdx.x;
  const int lane = tid & 31;
  const int wave = tid >> 5;
  const int bb   = blockIdx.y;
  const size_t qRow0 = (size_t)bb * SEQ + (size_t)blockIdx.x * 16;
  const size_t kRow0 = (size_t)bb * SEQ;
  const float scale = 0.03125f;                     // 1/sqrt(1024)

  v8f Oacc[8];
  for (int n = 0; n < 8; ++n) Oacc[n] = (v8f){};
  if (tid < 16) { mrow[tid] = -1e30f; lrow[tid] = 0.0f; }

  const int hi = (lane >> 4) << 3;                  // 0 or 8 (row-half select)

  for (int kbk = 0; kbk < SEQ; kbk += 32) {
    // ---- stage K block (32 keys x 1024 d, f16) into LDS ----
    for (int i = 0; i < 16; ++i) {
      const int idx = i * 256 + tid;                // 4096 16-byte chunks
      const int key = idx >> 7;                     // 0..31
      const int c8  = (idx & 127) << 3;             // half index, 16B aligned
      const _Float16* gsrc = &Kb[(kRow0 + kbk + key) * DMODEL + c8];
      _Float16*       ldst = &Ks[key][c8];
#if USE_ASYNC_LDS
      __builtin_amdgcn_global_load_async_to_lds_b128(
          (gv4i_p)gsrc, (lv4i_p)ldst, 0, 0);
#else
      *reinterpret_cast<v8h*>(ldst) = *reinterpret_cast<const v8h*>(gsrc);
#endif
    }
    // zero the score tile while the copies are in flight
    reinterpret_cast<float*>(Ss)[tid]       = 0.0f;
    reinterpret_cast<float*>(Ss)[256 + tid] = 0.0f;
    // prefetch next key block's V slice for this wave's columns
    if (kbk + 32 < SEQ) {
      const int pcol = wave * 128 + lane * 4;
      __builtin_prefetch(&VbT[((size_t)bb * DMODEL + pcol) * SEQ + kbk + 32], 0, 1);
    }
#if USE_ASYNC_LDS
    wait_async0();
#endif
    __syncthreads();                                // Ks + zeroed Ss visible

    // ---- partial scores: this wave's 128-wide d slice ----
    v8f sAcc0 = {}, sAcc1 = {};
    for (int c = 0; c < 4; ++c) {
      const int dd = wave * 128 + c * 32;
      const v16h a = frag_a(&Qb[(qRow0 + (lane & 15)) * DMODEL + dd], lane);
      {
        const v16h b = frag_b(&Ks[(lane & 15)][dd], lane);
        sAcc0 = __builtin_amdgcn_wmma_f32_16x16x32_f16(
            false, a, false, b, (short)0, sAcc0, false, false);
      }
      {
        const v16h b = frag_b(&Ks[16 + (lane & 15)][dd], lane);
        sAcc1 = __builtin_amdgcn_wmma_f32_16x16x32_f16(
            false, a, false, b, (short)0, sAcc1, false, false);
      }
    }
    // cross-wave reduction via LDS float atomics (ds_add_f32)
    for (int r = 0; r < 8; ++r) {
      atomicAdd(&Ss[hi + r][(lane & 15)],      sAcc0[r]);
      atomicAdd(&Ss[hi + r][16 + (lane & 15)], sAcc1[r]);
    }
    __syncthreads();

    // ---- online softmax update (one thread per query row) ----
    if (tid < 16) {
      const float mold = mrow[tid];
      float mnew = mold;
      for (int j = 0; j < 32; ++j)
        mnew = fmaxf(mnew, Ss[tid][j] * scale);
      const float alpha = __expf(mold - mnew);
      float ssum = 0.0f;
      for (int j = 0; j < 32; ++j) {
        const float p = __expf(Ss[tid][j] * scale - mnew);
        P[tid][j] = (_Float16)p;
        ssum += p;
      }
      mrow[tid] = mnew;
      lrow[tid] = lrow[tid] * alpha + ssum;
      arow[tid] = alpha;
    }
    __syncthreads();

    // ---- rescale accumulators, then P @ V for this wave's 128 columns ----
    float al[8];
    for (int r = 0; r < 8; ++r) al[r] = arow[hi + r];
    for (int n = 0; n < 8; ++n)
      for (int r = 0; r < 8; ++r) Oacc[n][r] *= al[r];

    const v16h ap = frag_a(&P[lane & 15][0], lane);
    for (int n = 0; n < 8; ++n) {
      const int col = wave * 128 + n * 16 + (lane & 15);
      const v16h b = frag_b(&VbT[((size_t)bb * DMODEL + col) * SEQ + kbk], lane);
      Oacc[n] = __builtin_amdgcn_wmma_f32_16x16x32_f16(
          false, ap, false, b, (short)0, Oacc[n], false, false);
    }
  }

  // ---- final normalization and fp32 store ----
  float invl[8];
  for (int r = 0; r < 8; ++r) invl[r] = 1.0f / lrow[hi + r];
  for (int n = 0; n < 8; ++n) {
    const int col = wave * 128 + n * 16 + (lane & 15);
    for (int r = 0; r < 8; ++r)
      out[(qRow0 + hi + r) * DMODEL + col] = Oacc[n][r] * invl[r];
  }
}

// ---------------------------------------------------------------------------
extern "C" void kernel_launch(void* const* d_in, const int* in_sizes, int n_in,
                              void* d_out, int out_size, void* d_ws, size_t ws_size,
                              hipStream_t stream) {
  const float* x  = (const float*)d_in[0];
  const float* Wq = (const float*)d_in[1];
  const float* bq = (const float*)d_in[2];
  const float* Wk = (const float*)d_in[3];
  const float* bk = (const float*)d_in[4];
  const float* Wv = (const float*)d_in[5];
  const float* bv = (const float*)d_in[6];
  float* out = (float*)d_out;

  // workspace: Q (16MB) | K (16MB) | V-transposed (16MB), all f16
  _Float16* Qb  = (_Float16*)d_ws;
  _Float16* Kb  = Qb + (size_t)NROWS * DMODEL;
  _Float16* VbT = Kb + (size_t)NROWS * DMODEL;

  dim3 g1(NROWS / 64, DMODEL / 64, 3);
  qkv_gemm_kernel<<<g1, 256, 0, stream>>>(x, Wq, bq, Wk, bk, Wv, bv,
                                          Qb, Kb, VbT);

  dim3 g2(SEQ / 16, NBATCH);
  flash_attn_kernel<<<g2, 256, 0, stream>>>(Qb, Kb, VbT, out);
}
